// Encoder_68401649156516
// MI455X (gfx1250) — matrix operand order repaired and checked by reading, not compiled, not run on hardware
//
#include <hip/hip_runtime.h>
#include <hip/hip_bf16.h>

// ---------------------------------------------------------------------------
// ViT encoder block on MI455X (gfx1250, wave32).
// All GEMMs run through v_wmma_f32_16x16x32_bf16 (bf16 operands, f32 acc).
// GEMM waves own 32(M) x 64(N) tiles: 8 WMMAs per 12 b128 loads per k-step.
// ---------------------------------------------------------------------------

typedef __attribute__((ext_vector_type(16))) __bf16 v16bf;
typedef __attribute__((ext_vector_type(8)))  float  v8f;

#define DEV_INLINE __device__ __forceinline__

constexpr int    D_    = 768;
constexpr int    H_    = 12;
constexpr int    HD_   = 64;
constexpr int    HID_  = 3072;
constexpr int    SEQ_  = 577;
constexpr int    BATCH_= 32;
constexpr size_t M_    = (size_t)BATCH_ * SEQ_;     // 18464 rows = 577 * 32

// -------------------------- workspace layout (bytes) -----------------------
constexpr size_t OFF_H    = 0;                                   // bf16 [M,768]
constexpr size_t SZ_H     = M_ * 768 * 2;                        // 28,360,704
constexpr size_t OFF_QKV  = OFF_H + SZ_H;                        // bf16 [M,2304]
constexpr size_t SZ_QKV   = M_ * 2304 * 2;                       // 85,082,112
constexpr size_t OFF_ATTN = OFF_QKV + SZ_QKV;                    // bf16 [M,768]
constexpr size_t OFF_HID  = OFF_QKV;                             // bf16 [M,3072]; aliases qkv+attn (dead by then)
constexpr size_t OFF_RES1 = OFF_QKV + M_ * 3072 * 2;             // f32 [M,768]
constexpr size_t SZ_RES1  = M_ * 768 * 4;
constexpr size_t OFF_WQKV = OFF_RES1 + SZ_RES1;                  // bf16 [2304,768] (transposed)
constexpr size_t OFF_WPROJ= OFF_WQKV + (size_t)768 * 2304 * 2;
constexpr size_t OFF_W1   = OFF_WPROJ + (size_t)768 * 768 * 2;
constexpr size_t OFF_W2   = OFF_W1 + (size_t)768 * 3072 * 2;
// total ws used: ~212.7 MB

// A-fragment k index for 16-bit 16x32 A per CDNA5 ISA:
// lanes 0-15 hold K {0..7,16..23}; lanes 16-31 hold K {8..15,24..31}.
DEV_INLINE int a_k(int e, int hi) { return ((e & 8) << 1) + (hi << 3) + (e & 7); }

DEV_INLINE float gelu_exact(float v) {
  return 0.5f * v * (1.0f + erff(v * 0.70710678118654752f));
}

// ---------------------------------------------------------------------------
// Weight convert + transpose: w[k,n] f32 -> wt[n,k] bf16 (K contiguous).
// ---------------------------------------------------------------------------
__global__ void conv_transpose_w(const float* __restrict__ w, __bf16* __restrict__ wt,
                                 int K, int N) {
  size_t i = (size_t)blockIdx.x * blockDim.x + threadIdx.x;
  size_t total = (size_t)K * N;
  if (i >= total) return;
  int k = (int)(i / N);
  int n = (int)(i % N);
  wt[(size_t)n * K + k] = (__bf16)w[i];
}

// ---------------------------------------------------------------------------
// LayerNorm (optionally two back-to-back LNs), one wave32 per row of 768.
// ---------------------------------------------------------------------------
template <bool DOUBLE>
__global__ __launch_bounds__(32)
void ln_rows(const float* __restrict__ x,
             const float* __restrict__ g1, const float* __restrict__ b1,
             const float* __restrict__ g2, const float* __restrict__ b2,
             __bf16* __restrict__ out) {
  const int lane = threadIdx.x;                // 0..31
  const size_t row = blockIdx.x;
  const float* xp = x + row * D_;
  float v[24];
  float s = 0.f;
#pragma unroll
  for (int i = 0; i < 24; ++i) { v[i] = xp[lane + (i << 5)]; s += v[i]; }
#pragma unroll
  for (int off = 16; off > 0; off >>= 1) s += __shfl_xor(s, off, 32);
  float mean = s * (1.0f / D_);
  float q = 0.f;
#pragma unroll
  for (int i = 0; i < 24; ++i) { float d = v[i] - mean; q += d * d; }
#pragma unroll
  for (int off = 16; off > 0; off >>= 1) q += __shfl_xor(q, off, 32);
  float rstd = rsqrtf(q * (1.0f / D_) + 1e-5f);
#pragma unroll
  for (int i = 0; i < 24; ++i)
    v[i] = (v[i] - mean) * rstd * g1[lane + (i << 5)] + b1[lane + (i << 5)];

  if (DOUBLE) {
    s = 0.f;
#pragma unroll
    for (int i = 0; i < 24; ++i) s += v[i];
#pragma unroll
    for (int off = 16; off > 0; off >>= 1) s += __shfl_xor(s, off, 32);
    mean = s * (1.0f / D_);
    q = 0.f;
#pragma unroll
    for (int i = 0; i < 24; ++i) { float d = v[i] - mean; q += d * d; }
#pragma unroll
    for (int off = 16; off > 0; off >>= 1) q += __shfl_xor(q, off, 32);
    rstd = rsqrtf(q * (1.0f / D_) + 1e-5f);
#pragma unroll
    for (int i = 0; i < 24; ++i)
      v[i] = (v[i] - mean) * rstd * g2[lane + (i << 5)] + b2[lane + (i << 5)];
  }
#pragma unroll
  for (int i = 0; i < 24; ++i) out[row * D_ + lane + (i << 5)] = (__bf16)v[i];
}

// ---------------------------------------------------------------------------
// bf16 WMMA GEMM: out[M,N] = A[M,K] @ BT[N,K]^T (+bias) (+epilogue)
//   EPI 0: store bf16
//   EPI 1: exact GELU, store bf16
//   EPI 2: += residual, store f32
// 4 waves / block; each wave owns a 32(M) x 64(N) tile (B frags reused 2x);
// K % 32 == 0, M % 32 == 0, N % 256 == 0.
// ---------------------------------------------------------------------------
template <int EPI>
__global__ __launch_bounds__(128)
void gemm_bf16_wmma(const __bf16* __restrict__ A, const __bf16* __restrict__ BT,
                    const float* __restrict__ bias, const float* __restrict__ res,
                    void* __restrict__ out, int N, int K) {
  const int lane = threadIdx.x & 31;
  const int wv   = threadIdx.x >> 5;
  const int hi   = lane >> 4;
  const int lm   = lane & 15;
  const int m0   = blockIdx.y << 5;                 // 32 rows per wave
  const int n0   = (blockIdx.x << 8) + (wv << 6);   // 64 cols per wave

  const __bf16* ap0 = A + (size_t)(m0 + lm) * K;
  const __bf16* ap1 = A + (size_t)(m0 + 16 + lm) * K;
  v8f acc[2][4] = {};

  for (int k0 = 0; k0 < K; k0 += 32) {
    __builtin_prefetch(ap0 + k0 + 64, 0, 1);       // global_prefetch_b8
    __builtin_prefetch(ap1 + k0 + 64, 0, 1);
    v16bf a0, a1;
#pragma unroll
    for (int e = 0; e < 16; ++e) {
      const int kk = k0 + (hi << 4) + e;
      a0[e] = ap0[kk];
      a1[e] = ap1[kk];
    }
#pragma unroll
    for (int t = 0; t < 4; ++t) {
      const __bf16* bp = BT + (size_t)(n0 + (t << 4) + lm) * K + k0 + (hi << 4);
      __builtin_prefetch(bp + 64, 0, 1);
      v16bf b;
#pragma unroll
      for (int e = 0; e < 16; ++e) b[e] = bp[e];
      acc[0][t] = __builtin_amdgcn_wmma_f32_16x16x32_bf16(
          false, a0, false, b, (short)0, acc[0][t], false, false);
      acc[1][t] = __builtin_amdgcn_wmma_f32_16x16x32_bf16(
          false, a1, false, b, (short)0, acc[1][t], false, false);
    }
  }

#pragma unroll
  for (int half = 0; half < 2; ++half) {
#pragma unroll
    for (int t = 0; t < 4; ++t) {
      const int col = n0 + (t << 4) + lm;
      const float bv = bias[col];
#pragma unroll
      for (int r = 0; r < 8; ++r) {
        const size_t idx = (size_t)(m0 + (half << 4) + r + (hi << 3)) * N + col;
        float v = acc[half][t][r] + bv;
        if (EPI == 1) v = gelu_exact(v);
        if (EPI == 2) {
          ((float*)out)[idx] = v + res[idx];
        } else {
          ((__bf16*)out)[idx] = (__bf16)v;
        }
      }
    }
  }
}

// ---------------------------------------------------------------------------
// Fused attention: one wave per (batch, head, 16-query-row tile).
// scores tile row-block staged in LDS, exact softmax (2 lanes per row,
// combined via shfl_xor 16), P@V via WMMA with zero-padded probs.
// ---------------------------------------------------------------------------
__global__ __launch_bounds__(32)
void attn_fused(const __bf16* __restrict__ qkv, __bf16* __restrict__ attn_out) {
  constexpr int NT   = 37;     // ceil(577/16) key tiles
  constexpr int NPAD = 608;    // 19 * 32, K-padding for PV WMMA
  __shared__ float  sc[16 * NPAD];
  __shared__ __bf16 pb[16 * NPAD];

  const int lane = threadIdx.x;        // 0..31
  const int hi   = lane >> 4;
  const int lm   = lane & 15;
  const int tile = blockIdx.x;
  const int head = blockIdx.y;
  const int b    = blockIdx.z;

  const size_t rowbase = (size_t)b * SEQ_;
  const __bf16* Qp = qkv + rowbase * 2304 + head * HD_;
  const __bf16* Kp = Qp + D_;
  const __bf16* Vp = Qp + 2 * D_;

  const int r0   = tile * 16;
  const int qrow = (r0 + lm < SEQ_) ? (r0 + lm) : (SEQ_ - 1);

  // Q as A-fragments (hd=64 -> two K=32 steps), loaded once.
  v16bf aq0, aq1;
  {
    const __bf16* qp = Qp + (size_t)qrow * 2304;
#pragma unroll
    for (int e = 0; e < 16; ++e) {
      const int kk = (hi << 4) + e;
      aq0[e] = qp[kk];
      aq1[e] = qp[32 + kk];
    }
  }

  // scores = (Q K^T) * 1/sqrt(64), one 16x16 tile per key tile -> LDS
  for (int j = 0; j < NT; ++j) {
    const int j0   = j * 16;
    const int krow = (j0 + lm < SEQ_) ? (j0 + lm) : (SEQ_ - 1);
    const __bf16* kp = Kp + (size_t)krow * 2304 + (hi << 4);
    v16bf b0, b1;
#pragma unroll
    for (int e = 0; e < 16; ++e) { b0[e] = kp[e]; b1[e] = kp[32 + e]; }
    v8f c = {};
    c = __builtin_amdgcn_wmma_f32_16x16x32_bf16(false, aq0, false, b0, (short)0, c, false, false);
    c = __builtin_amdgcn_wmma_f32_16x16x32_bf16(false, aq1, false, b1, (short)0, c, false, false);
#pragma unroll
    for (int r = 0; r < 8; ++r)
      sc[(r + (hi << 3)) * NPAD + j0 + lm] = c[r] * 0.125f;
  }
  __syncthreads();

  // exact softmax: 2 lanes per row (lane lm and lm+16 share row lm),
  // partials combined across the pair with shfl_xor(·, 16).
  {
    float mx = -3.0e38f;
    for (int c = hi; c < SEQ_; c += 2) mx = fmaxf(mx, sc[lm * NPAD + c]);
    mx = fmaxf(mx, __shfl_xor(mx, 16, 32));
    float s = 0.f;
    for (int c = hi; c < SEQ_; c += 2) {
      float e = __expf(sc[lm * NPAD + c] - mx);
      sc[lm * NPAD + c] = e;
      s += e;
    }
    s += __shfl_xor(s, 16, 32);
    const float inv = 1.f / s;
    for (int c = hi; c < NPAD; c += 2) {
      float pv = 0.f;
      if (c < SEQ_) pv = sc[lm * NPAD + c] * inv;
      pb[lm * NPAD + c] = (__bf16)pv;
    }
  }
  __syncthreads();

  // attn = P @ V, K runs over padded key dim (P==0 on padding)
  v8f o[4] = {};
  for (int k0 = 0; k0 < NPAD; k0 += 32) {
    v16bf a;
#pragma unroll
    for (int e = 0; e < 16; ++e) a[e] = pb[lm * NPAD + k0 + (hi << 4) + e];
#pragma unroll
    for (int t = 0; t < 4; ++t) {
      v16bf bv;
      const int kb = k0 + (hi << 4);
#pragma unroll
      for (int e = 0; e < 16; ++e) {
        const int krow = (kb + e < SEQ_) ? (kb + e) : (SEQ_ - 1);  // P==0 there anyway
        bv[e] = Vp[(size_t)krow * 2304 + (t << 4) + lm];
      }
      o[t] = __builtin_amdgcn_wmma_f32_16x16x32_bf16(
          false, a, false, bv, (short)0, o[t], false, false);
    }
  }

#pragma unroll
  for (int t = 0; t < 4; ++t) {
#pragma unroll
    for (int r = 0; r < 8; ++r) {
      const int mrow = r0 + r + (hi << 3);
      if (mrow < SEQ_)
        attn_out[(rowbase + mrow) * (size_t)D_ + head * HD_ + (t << 4) + lm] =
            (__bf16)o[t][r];
    }
  }
}

// ---------------------------------------------------------------------------
// Host-side launch
// ---------------------------------------------------------------------------
extern "C" void kernel_launch(void* const* d_in, const int* in_sizes, int n_in,
                              void* d_out, int out_size, void* d_ws, size_t ws_size,
                              hipStream_t stream) {
  const float* x     = (const float*)d_in[0];
  const float* wqkv  = (const float*)d_in[1];
  const float* bqkv  = (const float*)d_in[2];
  const float* wproj = (const float*)d_in[3];
  const float* bproj = (const float*)d_in[4];
  const float* g1    = (const float*)d_in[5];
  const float* be1   = (const float*)d_in[6];
  const float* g2    = (const float*)d_in[7];
  const float* be2   = (const float*)d_in[8];
  const float* g3    = (const float*)d_in[9];
  const float* be3   = (const float*)d_in[10];
  const float* w1    = (const float*)d_in[11];
  const float* bm1   = (const float*)d_in[12];
  const float* w2    = (const float*)d_in[13];
  const float* bm2   = (const float*)d_in[14];

  char* ws = (char*)d_ws;
  __bf16* h_bf    = (__bf16*)(ws + OFF_H);
  __bf16* qkv_bf  = (__bf16*)(ws + OFF_QKV);
  __bf16* attn_bf = (__bf16*)(ws + OFF_ATTN);
  __bf16* hid_bf  = (__bf16*)(ws + OFF_HID);
  float*  res1    = (float*)(ws + OFF_RES1);
  __bf16* wqkvT   = (__bf16*)(ws + OFF_WQKV);
  __bf16* wprojT  = (__bf16*)(ws + OFF_WPROJ);
  __bf16* w1T     = (__bf16*)(ws + OFF_W1);
  __bf16* w2T     = (__bf16*)(ws + OFF_W2);

  const int MT32 = (int)(M_ / 32);   // 577 row tiles of 32

  // 1) weights -> bf16, transposed ([N,K], K contiguous)
  conv_transpose_w<<<(768 * 2304 + 255) / 256, 256, 0, stream>>>(wqkv, wqkvT, 768, 2304);
  conv_transpose_w<<<(768 * 768 + 255) / 256, 256, 0, stream>>>(wproj, wprojT, 768, 768);
  conv_transpose_w<<<(768 * 3072 + 255) / 256, 256, 0, stream>>>(w1, w1T, 768, 3072);
  conv_transpose_w<<<(3072 * 768 + 255) / 256, 256, 0, stream>>>(w2, w2T, 3072, 768);

  // 2) LN1
  ln_rows<false><<<(int)M_, 32, 0, stream>>>(x, g1, be1, nullptr, nullptr, h_bf);

  // 3) QKV GEMM  [M,768] x [768,2304] + b
  gemm_bf16_wmma<0><<<dim3(2304 / 256, MT32), 128, 0, stream>>>(
      h_bf, wqkvT, bqkv, nullptr, qkv_bf, 2304, 768);

  // 4) fused attention
  attn_fused<<<dim3(37, H_, BATCH_), 32, 0, stream>>>(qkv_bf, attn_bf);

  // 5) proj GEMM + bias + residual(x) -> res1 (f32)
  gemm_bf16_wmma<2><<<dim3(768 / 256, MT32), 128, 0, stream>>>(
      attn_bf, wprojT, bproj, x, res1, 768, 768);

  // 6) LN2 then LN3 (fused)
  ln_rows<true><<<(int)M_, 32, 0, stream>>>(res1, g2, be2, g3, be3, h_bf);

  // 7) MLP up GEMM + bias + exact GELU -> hid (bf16)
  gemm_bf16_wmma<1><<<dim3(3072 / 256, MT32), 128, 0, stream>>>(
      h_bf, w1T, bm1, nullptr, hid_bf, 3072, 768);

  // 8) MLP down GEMM + bias + residual(res1) -> d_out (f32)
  gemm_bf16_wmma<2><<<dim3(768 / 256, MT32), 128, 0, stream>>>(
      hid_bf, w2T, bm2, res1, (float*)d_out, 768, 3072);
}
